// CrossAttn_71184787964286
// MI455X (gfx1250) — matrix operand order repaired and checked by reading, not compiled, not run on hardware
//
#include <hip/hip_runtime.h>

// Fused cross-attention pipeline for MI455X (gfx1250, wave32, WMMA).
//   k0: one-shot f32->f16 conversion of iQ, iK, Wq, Wkv, Wo into workspace
//       (bandwidth-trivial; removes all v_cvt from the GEMM hot loops and
//        halves K-loop load bytes)
//   k1: rq  = iQh @ Wqh.T           (pure f16 WMMA)
//   k2: k,v = iKh @ Wkvh.T          (K row-major, V transposed (B,H,AD,S))
//   k3: relu^2-normalized attention, streaming over S (linear norm -> no
//       max-rescaling; 8 WMMAs per 32-wide S step per wave)
//   k4: out = o @ Woh.T             (f16 WMMA -> f32 out)
// Workspace layout (MiB): rq[0,8) rk[8,16) rvT[16,24) o[24,32)
//                         iQh[32,40) iKh[40,48) Wqh[48,50) Wkvh[50,54) Woh[54,56)

typedef __attribute__((ext_vector_type(16))) _Float16 v16h;
typedef __attribute__((ext_vector_type(8)))  _Float16 v8h;
typedef __attribute__((ext_vector_type(8)))  float    v8f;

constexpr int   kB  = 2, kQ = 2048, kS = 2048, kD = 1024, kH = 16, kAD = 64;
constexpr float kScale = 0.125f;   // 1/sqrt(64)
constexpr float kEps   = 1e-32f;

__device__ __forceinline__ v8f wmma16(v16h a, v16h b, v8f c) {
  return __builtin_amdgcn_wmma_f32_16x16x32_f16(false, a, false, b, (short)0, c,
                                                false, false);
}

// A-fragment (16x32, MxK) from row-major f16. p = &A[row*ld + k0 + half*16].
// Per-lane v16h order: K = {off..off+15}.
__device__ __forceinline__ v16h a_frag_f16(const _Float16* p) {
  v8h lo = *(const v8h*)p;
  v8h hi = *(const v8h*)(p + 8);
  v16h r;
#pragma unroll
  for (int i = 0; i < 8; ++i) { r[i] = lo[i]; r[i + 8] = hi[i]; }
  return r;
}

// B-fragment (32x16, KxN), B[k][n] = W[n][k]: per-lane v16h = 16 consecutive
// K values at &W[n*ld + k0 + half*16] -> a single 32B contiguous load.
__device__ __forceinline__ v16h b_frag_f16(const _Float16* p) {
  return *(const v16h*)p;
}

// ---------------- k0: bulk f32 -> f16 (8 elements / thread) -------------------
__global__ __launch_bounds__(256) void cvt_f16_kernel(const float* __restrict__ src,
                                                      _Float16* __restrict__ dst,
                                                      int n8) {
  const int i = blockIdx.x * blockDim.x + threadIdx.x;
  if (i >= n8) return;
  const float4* p = (const float4*)src + (size_t)i * 2;
  const float4 a = p[0], b = p[1];
  v8h r;
  r[0] = (_Float16)a.x; r[1] = (_Float16)a.y; r[2] = (_Float16)a.z; r[3] = (_Float16)a.w;
  r[4] = (_Float16)b.x; r[5] = (_Float16)b.y; r[6] = (_Float16)b.z; r[7] = (_Float16)b.w;
  ((v8h*)dst)[i] = r;
}

// ---------------- k1: rq = iQh @ Wqh.T  (M=4096, N=1024, K=1024) --------------
__global__ __launch_bounds__(128) void proj_q_kernel(const _Float16* __restrict__ A,
                                                     const _Float16* __restrict__ W,
                                                     _Float16* __restrict__ C) {
  const int lane = threadIdx.x & 31, wv = threadIdx.x >> 5;
  const int half = lane >> 4, l15 = lane & 15;
  const int row0 = (blockIdx.x * 4 + wv) * 16;
  const int col0 = blockIdx.y * 64;

  const _Float16* arow = A + (size_t)(row0 + l15) * kD + half * 16;
  const _Float16* w0 = W + (size_t)(col0 +  0 + l15) * kD + half * 16;
  const _Float16* w1 = W + (size_t)(col0 + 16 + l15) * kD + half * 16;
  const _Float16* w2 = W + (size_t)(col0 + 32 + l15) * kD + half * 16;
  const _Float16* w3 = W + (size_t)(col0 + 48 + l15) * kD + half * 16;

  v8f c0 = {}, c1 = {}, c2 = {}, c3 = {};
  for (int k0 = 0; k0 < kD; k0 += 32) {
    const v16h a = a_frag_f16(arow + k0);
    c0 = wmma16(a, b_frag_f16(w0 + k0), c0);
    c1 = wmma16(a, b_frag_f16(w1 + k0), c1);
    c2 = wmma16(a, b_frag_f16(w2 + k0), c2);
    c3 = wmma16(a, b_frag_f16(w3 + k0), c3);
  }
#pragma unroll
  for (int r = 0; r < 8; ++r) {
    _Float16* out = C + (size_t)(row0 + r + half * 8) * kD + col0 + l15;
    out[0]  = (_Float16)c0[r];
    out[16] = (_Float16)c1[r];
    out[32] = (_Float16)c2[r];
    out[48] = (_Float16)c3[r];
  }
}

// ------------- k2: kv = iKh @ Wkvh.T  (M=4096, N=2048, K=1024) ----------------
// cols [0,1024) -> rk row-major (B*S, D);  cols [1024,2048) -> rvT (B,H,AD,S)
__global__ __launch_bounds__(128) void proj_kv_kernel(const _Float16* __restrict__ A,
                                                      const _Float16* __restrict__ W,
                                                      _Float16* __restrict__ RK,
                                                      _Float16* __restrict__ RVT) {
  const int lane = threadIdx.x & 31, wv = threadIdx.x >> 5;
  const int half = lane >> 4, l15 = lane & 15;
  const int row0 = (blockIdx.x * 4 + wv) * 16;
  const int col0 = blockIdx.y * 64;

  const _Float16* arow = A + (size_t)(row0 + l15) * kD + half * 16;
  const _Float16* w0 = W + (size_t)(col0 +  0 + l15) * kD + half * 16;
  const _Float16* w1 = W + (size_t)(col0 + 16 + l15) * kD + half * 16;
  const _Float16* w2 = W + (size_t)(col0 + 32 + l15) * kD + half * 16;
  const _Float16* w3 = W + (size_t)(col0 + 48 + l15) * kD + half * 16;

  v8f c0 = {}, c1 = {}, c2 = {}, c3 = {};
  for (int k0 = 0; k0 < kD; k0 += 32) {
    const v16h a = a_frag_f16(arow + k0);
    c0 = wmma16(a, b_frag_f16(w0 + k0), c0);
    c1 = wmma16(a, b_frag_f16(w1 + k0), c1);
    c2 = wmma16(a, b_frag_f16(w2 + k0), c2);
    c3 = wmma16(a, b_frag_f16(w3 + k0), c3);
  }
  if (col0 < kD) {  // K heads: natural row-major
#pragma unroll
    for (int r = 0; r < 8; ++r) {
      _Float16* out = RK + (size_t)(row0 + r + half * 8) * kD + col0 + l15;
      out[0]  = (_Float16)c0[r];
      out[16] = (_Float16)c1[r];
      out[32] = (_Float16)c2[r];
      out[48] = (_Float16)c3[r];
    }
  } else {          // V heads: transpose-on-store into (B,H,AD,S)
#pragma unroll
    for (int r = 0; r < 8; ++r) {
      const int m = row0 + r + half * 8;
      const int b = m / kS, s = m % kS;
      const int nb = col0 - kD + l15;
      {
        const int n = nb +  0, h = n >> 6, d = n & 63;
        RVT[(((size_t)b * kH + h) * kAD + d) * kS + s] = (_Float16)c0[r];
      }
      {
        const int n = nb + 16, h = n >> 6, d = n & 63;
        RVT[(((size_t)b * kH + h) * kAD + d) * kS + s] = (_Float16)c1[r];
      }
      {
        const int n = nb + 32, h = n >> 6, d = n & 63;
        RVT[(((size_t)b * kH + h) * kAD + d) * kS + s] = (_Float16)c2[r];
      }
      {
        const int n = nb + 48, h = n >> 6, d = n & 63;
        RVT[(((size_t)b * kH + h) * kAD + d) * kS + s] = (_Float16)c3[r];
      }
    }
  }
}

// ---------------- k3: streaming relu^2-normalized attention -------------------
// One wave per (b, h, 16-row q-tile). 8 WMMAs per 32-wide S step.
__global__ __launch_bounds__(128) void attn_kernel(const _Float16* __restrict__ rq,
                                                   const _Float16* __restrict__ rk,
                                                   const _Float16* __restrict__ rvT,
                                                   const unsigned char* __restrict__ mask,
                                                   const float* __restrict__ nbias,
                                                   _Float16* __restrict__ o) {
  __shared__ _Float16 tbuf[4 * 16 * 32];  // per-wave 16x32 t-tile slices
  const int lane = threadIdx.x & 31, wv = threadIdx.x >> 5;
  const int half = lane >> 4, l15 = lane & 15;
  const int wid = blockIdx.x * 4 + wv;
  const int qt = wid % (kQ / 16);
  const int bh = wid / (kQ / 16);
  const int h = bh % kH, b = bh / kH;
  _Float16* tl = tbuf + wv * 512;
  const float nb = nbias[0];

  // Q A-fragments: invariant over the S loop (K = ADIM = 64 -> two fragments)
  const _Float16* qrow =
      rq + (size_t)(b * kQ + qt * 16 + l15) * kD + h * kAD + half * 16;
  const v16h aq0 = a_frag_f16(qrow);
  const v16h aq1 = a_frag_f16(qrow + 32);

  v8f acc0 = {}, acc1 = {}, acc2 = {}, acc3 = {};
  float dp[8] = {0.f, 0.f, 0.f, 0.f, 0.f, 0.f, 0.f, 0.f};

  const size_t mbase = ((size_t)b * kQ + qt * 16) * kS;
  const _Float16* kbase = rk + (size_t)b * kS * kD + h * kAD;
  const _Float16* vbase = rvT + ((size_t)b * kH + h) * kAD * kS;

  for (int s0 = 0; s0 < kS; s0 += 32) {
    // scores -> t for two 16-wide s-subtiles
#pragma unroll
    for (int sub = 0; sub < 2; ++sub) {
      const int srow = s0 + sub * 16 + l15;
      const _Float16* kr = kbase + (size_t)srow * kD + half * 16;
      v8f sc = {};
      sc = wmma16(aq0, *(const v16h*)kr, sc);
      sc = wmma16(aq1, *(const v16h*)(kr + 32), sc);
#pragma unroll
      for (int r = 0; r < 8; ++r) {
        const int row = r + half * 8;
        const unsigned char mk = mask[mbase + (size_t)row * kS + srow];
        float x = sc[r] * kScale + nb;
        x = (x > 0.f && !mk) ? x : 0.f;
        const float t = x * x;
        dp[r] += t;
        tl[row * 32 + sub * 16 + l15] = (_Float16)t;
      }
    }
    // re-fragment t (C layout) as an A matrix via the per-wave LDS slice
    asm volatile("s_wait_dscnt 0" ::: "memory");
    const v16h ta = a_frag_f16(tl + l15 * 32 + half * 16);

    // acc += t @ rv  (B-frags contiguous thanks to transposed V)
    const _Float16* vr = vbase + s0 + half * 16;
    acc0 = wmma16(ta, *(const v16h*)(vr + (size_t)( 0 + l15) * kS), acc0);
    acc1 = wmma16(ta, *(const v16h*)(vr + (size_t)(16 + l15) * kS), acc1);
    acc2 = wmma16(ta, *(const v16h*)(vr + (size_t)(32 + l15) * kS), acc2);
    acc3 = wmma16(ta, *(const v16h*)(vr + (size_t)(48 + l15) * kS), acc3);
  }

  // Row-sum denominator: reduce across the 16-lane group that shares rows
#pragma unroll
  for (int r = 0; r < 8; ++r) {
    float v = dp[r];
    v += __shfl_xor(v, 1, 16);
    v += __shfl_xor(v, 2, 16);
    v += __shfl_xor(v, 4, 16);
    v += __shfl_xor(v, 8, 16);
    dp[r] = v;
  }

  _Float16* orow = o + ((size_t)b * kQ + qt * 16) * kD + h * kAD;
#pragma unroll
  for (int r = 0; r < 8; ++r) {
    const int row = r + half * 8;
    const float inv = 1.f / (dp[r] + kEps);
    _Float16* op = orow + (size_t)row * kD + l15;
    op[0]  = (_Float16)(acc0[r] * inv);
    op[16] = (_Float16)(acc1[r] * inv);
    op[32] = (_Float16)(acc2[r] * inv);
    op[48] = (_Float16)(acc3[r] * inv);
  }
}

// ---------------- k4: out = o @ Woh.T  (M=4096, N=1024, K=1024) ---------------
__global__ __launch_bounds__(128) void outproj_kernel(const _Float16* __restrict__ A,
                                                      const _Float16* __restrict__ W,
                                                      float* __restrict__ out) {
  const int lane = threadIdx.x & 31, wv = threadIdx.x >> 5;
  const int half = lane >> 4, l15 = lane & 15;
  const int row0 = (blockIdx.x * 4 + wv) * 16;
  const int col0 = blockIdx.y * 64;

  const _Float16* arow = A + (size_t)(row0 + l15) * kD + half * 16;
  const _Float16* w0 = W + (size_t)(col0 +  0 + l15) * kD + half * 16;
  const _Float16* w1 = W + (size_t)(col0 + 16 + l15) * kD + half * 16;
  const _Float16* w2 = W + (size_t)(col0 + 32 + l15) * kD + half * 16;
  const _Float16* w3 = W + (size_t)(col0 + 48 + l15) * kD + half * 16;

  v8f c0 = {}, c1 = {}, c2 = {}, c3 = {};
  for (int k0 = 0; k0 < kD; k0 += 32) {
    const v16h a = a_frag_f16(arow + k0);
    c0 = wmma16(a, b_frag_f16(w0 + k0), c0);
    c1 = wmma16(a, b_frag_f16(w1 + k0), c1);
    c2 = wmma16(a, b_frag_f16(w2 + k0), c2);
    c3 = wmma16(a, b_frag_f16(w3 + k0), c3);
  }
#pragma unroll
  for (int r = 0; r < 8; ++r) {
    float* op = out + (size_t)(row0 + r + half * 8) * kD + col0 + l15;
    op[0]  = c0[r];
    op[16] = c1[r];
    op[32] = c2[r];
    op[48] = c3[r];
  }
}

extern "C" void kernel_launch(void* const* d_in, const int* in_sizes, int n_in,
                              void* d_out, int out_size, void* d_ws, size_t ws_size,
                              hipStream_t stream) {
  const float*         iQ    = (const float*)d_in[0];
  const float*         iK    = (const float*)d_in[1];
  const unsigned char* mask  = (const unsigned char*)d_in[2];  // bool tensor
  const float*         Wq    = (const float*)d_in[3];
  const float*         Wkv   = (const float*)d_in[4];
  const float*         Wo    = (const float*)d_in[5];
  const float*         nbias = (const float*)d_in[6];
  float*               out   = (float*)d_out;

  char* ws = (char*)d_ws;
  _Float16* rq   = (_Float16*)(ws);
  _Float16* rk   = (_Float16*)(ws + ((size_t)8 << 20));
  _Float16* rvT  = (_Float16*)(ws + ((size_t)16 << 20));
  _Float16* o    = (_Float16*)(ws + ((size_t)24 << 20));
  _Float16* iQh  = (_Float16*)(ws + ((size_t)32 << 20));
  _Float16* iKh  = (_Float16*)(ws + ((size_t)40 << 20));
  _Float16* Wqh  = (_Float16*)(ws + ((size_t)48 << 20));
  _Float16* Wkvh = (_Float16*)(ws + ((size_t)50 << 20));
  _Float16* Woh  = (_Float16*)(ws + ((size_t)54 << 20));

  const dim3 blk(128);  // 4 waves / workgroup

  // k0: one-shot f32 -> f16 conversions (n/8 threads each; all sizes % 8 == 0)
  {
    const int nQ  = kB * kQ * kD / 8;   // 524288
    const int nK  = kB * kS * kD / 8;   // 524288
    const int nWq = kD * kD / 8;        // 131072
    const int nWk = 2 * kD * kD / 8;    // 262144
    const int nWo = kD * kD / 8;        // 131072
    cvt_f16_kernel<<<dim3(nQ  / 256), dim3(256), 0, stream>>>(iQ,  iQh,  nQ);
    cvt_f16_kernel<<<dim3(nK  / 256), dim3(256), 0, stream>>>(iK,  iKh,  nK);
    cvt_f16_kernel<<<dim3(nWq / 256), dim3(256), 0, stream>>>(Wq,  Wqh,  nWq);
    cvt_f16_kernel<<<dim3(nWk / 256), dim3(256), 0, stream>>>(Wkv, Wkvh, nWk);
    cvt_f16_kernel<<<dim3(nWo / 256), dim3(256), 0, stream>>>(Wo,  Woh,  nWo);
  }

  proj_q_kernel  <<<dim3(64, 16), blk, 0, stream>>>(iQh, Wqh, rq);
  proj_kv_kernel <<<dim3(64, 32), blk, 0, stream>>>(iKh, Wkvh, rk, rvT);
  attn_kernel    <<<dim3(1024),   blk, 0, stream>>>(rq, rk, rvT, mask, nbias, o);
  outproj_kernel <<<dim3(64, 16), blk, 0, stream>>>(o, Woh, out);
  (void)in_sizes; (void)n_in; (void)out_size; (void)ws_size;
}
